// SparseMultiheadAttention_65962107732186
// MI455X (gfx1250) — matrix operand order repaired and checked by reading, not compiled, not run on hardware
//
#include <hip/hip_runtime.h>

// ---------------------------------------------------------------------------
// Sparse multi-head attention for MI455X (gfx1250, wave32, WMMA).
// B=4, S=2048, D_MODEL=1024, H=16, d_k=64.  All GEMM work (projections,
// QK^T, PV, output projection) runs on v_wmma_f32_16x16x32_bf16.
// Staging uses GLOBAL_LOAD_ASYNC_TO_LDS (ASYNCcnt) with double buffering
// when the builtin is available; falls back to load+ds_store otherwise.
// ---------------------------------------------------------------------------

typedef __attribute__((ext_vector_type(16))) __bf16 v16bf;
typedef __attribute__((ext_vector_type(8)))  float  v8f;
typedef __attribute__((ext_vector_type(4)))  int    v4i;

#define S_LEN  2048
#define DMODEL 1024
#define NHEAD  16
#define DK     64
#define MROWS  8192            // B*S
#define BHEADS 64              // B*H

#if defined(__has_builtin)
#  if __has_builtin(__builtin_amdgcn_global_load_async_to_lds_b128) && \
      __has_builtin(__builtin_amdgcn_s_wait_asynccnt)
#    define USE_ASYNC_LDS 1
#  endif
#endif
#ifndef USE_ASYNC_LDS
#  define USE_ASYNC_LDS 0
#endif

#if USE_ASYNC_LDS
#define AS_GLOBAL __attribute__((address_space(1)))
#define AS_LOCAL  __attribute__((address_space(3)))
// async 16B copy global -> LDS (per-lane addresses), tracked by ASYNCcnt
__device__ __forceinline__ void async_cp16(const unsigned short* g, unsigned short* l) {
  __builtin_amdgcn_global_load_async_to_lds_b128(
      (AS_GLOBAL v4i*)(AS_GLOBAL void*)(g),
      (AS_LOCAL  v4i*)(AS_LOCAL  void*)(l), 0, 0);
}
__device__ __forceinline__ void async_wait0() {
  __builtin_amdgcn_s_wait_asynccnt(0);
}
#else
__device__ __forceinline__ void async_wait0() {}
#endif

__device__ __forceinline__ unsigned short f32_to_bf16(float f) {
  union { float f; unsigned u; } v; v.f = f;
  unsigned r = v.u + 0x7FFFu + ((v.u >> 16) & 1u);   // round-nearest-even
  return (unsigned short)(r >> 16);
}

// deterministic stand-in for jax.random.uniform (threefry not replicable here)
__device__ __forceinline__ float rng_uniform(unsigned idx, unsigned salt) {
  unsigned a = idx * 0x9E3779B9u + salt * 0x85EBCA6Bu;
  a ^= a >> 16; a *= 0x7feb352du;
  a ^= a >> 15; a *= 0x846ca68bu;
  a ^= a >> 16;
  return (float)(a >> 8) * (1.0f / 16777216.0f);
}

__device__ __forceinline__ v8f wmma_bf16(v16bf a, v16bf b, v8f c) {
  return __builtin_amdgcn_wmma_f32_16x16x32_bf16(
      /*neg_a=*/false, a, /*neg_b=*/false, b,
      /*c_mod=*/(short)0, c, /*reuse_a=*/false, /*reuse_b=*/false);
}

// A operand (16x32, row-major source, stride ld halves).
// ISA layout: lane m = l%16; k chunks [base, base+8) and [base+16, base+24),
// base = (l/16)*8.
__device__ __forceinline__ v16bf load_A(const unsigned short* base, int ld, int lane) {
  int m  = lane & 15;
  int kb = (lane >> 4) * 8;
  const unsigned short* p = base + m * ld + kb;
  union { v16bf v; uint4 q[2]; } u;
  u.q[0] = *(const uint4*)(p);
  u.q[1] = *(const uint4*)(p + 16);
  return u.v;
}

// B operand (32x16) from an "n-major" source: src[n][k] contiguous in k.
// ISA layout: lane n = l%16; k chunks [base, base+8) and [base+16, base+24),
// base = (l/16)*8.
__device__ __forceinline__ v16bf load_Bt(const unsigned short* base, int ld, int lane) {
  int n  = lane & 15;
  int kb = (lane >> 4) * 8;
  const unsigned short* p = base + n * ld + kb;
  union { v16bf v; uint4 q[2]; } u;
  u.q[0] = *(const uint4*)(p);
  u.q[1] = *(const uint4*)(p + 16);
  return u.v;
}

// ---------------------------------------------------------------------------
// Kernel 1: fp32 -> bf16 conversion
// ---------------------------------------------------------------------------
__global__ void cvt_bf16(const float* __restrict__ in,
                         unsigned short* __restrict__ out, int n) {
  for (int i = blockIdx.x * blockDim.x + threadIdx.x; i < n;
       i += gridDim.x * blockDim.x)
    out[i] = f32_to_bf16(in[i]);
}

// ---------------------------------------------------------------------------
// Kernel 2: tiled GEMM  C[M,N] = X[M,K] @ W[N,K]^T + bias
//   mode 0: write bf16, permuted to [B,H,S,DK]  (QKV projections; scale folds
//           1/sqrt(dk) for Q);  mode 1: write f32 row-major [M,N] to d_out.
// Block tile 128x64, K chunk 64, double-buffered async LDS staging,
// 8 waves (each 16 rows x 64 cols = 4 accumulators, 8 WMMAs per K chunk).
// ---------------------------------------------------------------------------
#define GM_MBLK 128
#define GM_NBLK 64
#define GM_KC   64

__device__ __forceinline__ void gemm_stage(
    const unsigned short* __restrict__ X, const unsigned short* __restrict__ W,
    unsigned short* Xs, unsigned short* Ws_,
    int bm, int bn, int K, int k0, int t) {
#if USE_ASYNC_LDS
#pragma unroll
  for (int i = 0; i < 4; ++i) {            // 128 rows x 8 segs = 1024 x 16B
    int seg = t + i * 256;
    int row = seg >> 3, sc = (seg & 7) * 8;
    async_cp16(&X[(size_t)(bm + row) * K + k0 + sc], &Xs[row * GM_KC + sc]);
  }
#pragma unroll
  for (int i = 0; i < 2; ++i) {            // 64 rows x 8 segs = 512 x 16B
    int seg = t + i * 256;
    int row = seg >> 3, sc = (seg & 7) * 8;
    async_cp16(&W[(size_t)(bn + row) * K + k0 + sc], &Ws_[row * GM_KC + sc]);
  }
#else
#pragma unroll
  for (int i = 0; i < 4; ++i) {
    int seg = t + i * 256;
    int row = seg >> 3, sc = (seg & 7) * 8;
    *(uint4*)&Xs[row * GM_KC + sc] =
        *(const uint4*)&X[(size_t)(bm + row) * K + k0 + sc];
  }
#pragma unroll
  for (int i = 0; i < 2; ++i) {
    int seg = t + i * 256;
    int row = seg >> 3, sc = (seg & 7) * 8;
    *(uint4*)&Ws_[row * GM_KC + sc] =
        *(const uint4*)&W[(size_t)(bn + row) * K + k0 + sc];
  }
  if (k0 + GM_KC < K)  // gfx1250 global_prefetch_b8 hint for next K chunk
    __builtin_prefetch(&X[(size_t)(bm + (t >> 1)) * K + k0 + GM_KC], 0, 1);
#endif
}

__global__ __launch_bounds__(256) void gemm_bf16(
    const unsigned short* __restrict__ X,   // [M][K] bf16
    const unsigned short* __restrict__ W,   // [N][K] bf16 (row-major == B^T)
    const float* __restrict__ bias,         // [N]
    unsigned short* __restrict__ outBf,     // mode 0
    float* __restrict__ outF,               // mode 1
    int M, int N, int K, float scale, int mode) {
  __shared__ __align__(16) unsigned short Xs[2][GM_MBLK * GM_KC];
  __shared__ __align__(16) unsigned short Ws[2][GM_NBLK * GM_KC];

  const int bm = blockIdx.x * GM_MBLK;
  const int bn = blockIdx.y * GM_NBLK;
  const int t = threadIdx.x, wave = t >> 5, lane = t & 31;

  v8f acc[4];
#pragma unroll
  for (int i = 0; i < 4; ++i)
#pragma unroll
    for (int j = 0; j < 8; ++j) acc[i][j] = 0.f;

  gemm_stage(X, W, Xs[0], Ws[0], bm, bn, K, 0, t);
  async_wait0();
  __syncthreads();

  const int nk = K / GM_KC;
  for (int it = 0; it < nk; ++it) {
    const int cur = it & 1;
    if (it + 1 < nk)  // prefetch next K chunk into the other buffer
      gemm_stage(X, W, Xs[cur ^ 1], Ws[cur ^ 1], bm, bn, K, (it + 1) * GM_KC, t);

    const unsigned short* xs = Xs[cur];
    const unsigned short* ws = Ws[cur];
    v16bf a0 = load_A(xs + (wave * 16) * GM_KC, GM_KC, lane);
    v16bf a1 = load_A(xs + (wave * 16) * GM_KC + 32, GM_KC, lane);
#pragma unroll
    for (int tn = 0; tn < 4; ++tn) {
      v16bf b0 = load_Bt(ws + (tn * 16) * GM_KC, GM_KC, lane);
      v16bf b1 = load_Bt(ws + (tn * 16) * GM_KC + 32, GM_KC, lane);
      acc[tn] = wmma_bf16(a0, b0, acc[tn]);
      acc[tn] = wmma_bf16(a1, b1, acc[tn]);
    }
    async_wait0();
    __syncthreads();
  }

  // epilogue: C element (row = j + 8*(lane/16), col = lane%16) per tile
  const int rbase = bm + wave * 16 + ((lane >> 4) * 8);
#pragma unroll
  for (int tn = 0; tn < 4; ++tn) {
    int n = bn + tn * 16 + (lane & 15);
    float bv = bias[n];
#pragma unroll
    for (int j = 0; j < 8; ++j) {
      int m = rbase + j;
      float v = (acc[tn][j] + bv) * scale;
      if (mode == 0) {
        int b_ = m >> 11, s_ = m & (S_LEN - 1);
        int h_ = n >> 6, d_ = n & (DK - 1);
        outBf[((size_t)(b_ * NHEAD + h_) * S_LEN + s_) * DK + d_] = f32_to_bf16(v);
      } else {
        outF[(size_t)m * N + n] = v;
      }
    }
  }
}

// ---------------------------------------------------------------------------
// Kernel 3: flash-style sparse attention with online softmax + dropout
// grid = (B*H, S/128); block = 256 threads (8 waves, 16 q-rows each)
// ---------------------------------------------------------------------------
#define AT_KC 32
#define VTS   40  // padded LDS stride for transposed V

__global__ __launch_bounds__(256) void attn_kernel(
    const unsigned short* __restrict__ Qp,  // [BH][S][DK] bf16 (pre-scaled 1/8)
    const unsigned short* __restrict__ Kp,
    const unsigned short* __restrict__ Vp,
    unsigned short* __restrict__ AO) {      // [B][S][DMODEL] bf16
  __shared__ __align__(16) unsigned short Ks[AT_KC * DK];   // [kv][d]
  __shared__ __align__(16) unsigned short Vt[DK * VTS];     // [d][kv] transposed
  __shared__ __align__(16) unsigned short Ps[8][16 * AT_KC];// per-wave P scratch

  const int bh = blockIdx.x;
  const int qb = blockIdx.y * 128;
  const int t = threadIdx.x, wave = t >> 5, lane = t & 31;
  const int q0 = qb + wave * 16;

  const unsigned short* Qb = Qp + (size_t)bh * S_LEN * DK;
  const unsigned short* Kb = Kp + (size_t)bh * S_LEN * DK;
  const unsigned short* Vb = Vp + (size_t)bh * S_LEN * DK;

  // Q tile 16x64 in A layout (two 16x32 operands)
  v16bf qa0 = load_A(Qb + (size_t)q0 * DK, DK, lane);
  v16bf qa1 = load_A(Qb + (size_t)q0 * DK + 32, DK, lane);

  const float NEG_INF = -__builtin_inff();
  v8f o[4];
  float mrow[8], lrow[8];
#pragma unroll
  for (int i = 0; i < 4; ++i)
#pragma unroll
    for (int j = 0; j < 8; ++j) o[i][j] = 0.f;
#pragma unroll
  for (int j = 0; j < 8; ++j) { mrow[j] = NEG_INF; lrow[j] = 0.f; }

  for (int kc = 0; kc < S_LEN; kc += AT_KC) {
    __syncthreads();
    {  // stage K rows (async) + V transposed (through VGPRs)
      int row = t >> 3, sc = (t & 7) * 8;
#if USE_ASYNC_LDS
      async_cp16(&Kb[(size_t)(kc + row) * DK + sc], &Ks[row * DK + sc]);
#else
      *(uint4*)&Ks[row * DK + sc] = *(const uint4*)&Kb[(size_t)(kc + row) * DK + sc];
#endif
      uint4 vq = *(const uint4*)&Vb[(size_t)(kc + row) * DK + sc];
      const unsigned short* pv = (const unsigned short*)&vq;
#pragma unroll
      for (int i = 0; i < 8; ++i) Vt[(sc + i) * VTS + row] = pv[i];
    }
    async_wait0();
    __syncthreads();

    // scores: two 16x16 tiles (kv cols kc..kc+15 and kc+16..kc+31)
    v8f sA, sB;
#pragma unroll
    for (int j = 0; j < 8; ++j) { sA[j] = 0.f; sB[j] = 0.f; }
    sA = wmma_bf16(qa0, load_Bt(Ks + 0 * DK, DK, lane), sA);
    sA = wmma_bf16(qa1, load_Bt(Ks + 0 * DK + 32, DK, lane), sA);
    sB = wmma_bf16(qa0, load_Bt(Ks + 16 * DK, DK, lane), sB);
    sB = wmma_bf16(qa1, load_Bt(Ks + 16 * DK + 32, DK, lane), sB);

    // sparsity mask + online softmax + dropout; row j lives in one 16-lane half
#pragma unroll
    for (int j = 0; j < 8; ++j) {
      int row = j + 8 * (lane >> 4);
      unsigned qg = (unsigned)(q0 + row);
      unsigned c0 = (unsigned)(kc + (lane & 15));
      unsigned i0 = ((unsigned)bh * S_LEN + qg) * S_LEN + c0;
      unsigned i1 = i0 + 16;
      float s0 = sA[j], s1 = sB[j];
      if (rng_uniform(i0, 42u) <= 0.9f) s0 = NEG_INF;   // keep where u > sparsity
      if (rng_uniform(i1, 42u) <= 0.9f) s1 = NEG_INF;

      float cm = fmaxf(s0, s1);
#pragma unroll
      for (int off = 1; off < 16; off <<= 1)
        cm = fmaxf(cm, __shfl_xor(cm, off, 32));
      float mn = fmaxf(mrow[j], cm);
      float rs = (mn == mrow[j]) ? 1.f : __expf(mrow[j] - mn);

      float p0 = (s0 == NEG_INF) ? 0.f : __expf(s0 - mn);
      float p1 = (s1 == NEG_INF) ? 0.f : __expf(s1 - mn);
      float ps = p0 + p1;
#pragma unroll
      for (int off = 1; off < 16; off <<= 1)
        ps += __shfl_xor(ps, off, 32);
      lrow[j] = lrow[j] * rs + ps;     // denominator ignores dropout
      mrow[j] = mn;
#pragma unroll
      for (int tn = 0; tn < 4; ++tn) o[tn][j] *= rs;

      // dropout on numerator terms (bernoulli keep=0.9)
      if (rng_uniform(i0, 7u) >= 0.9f) p0 = 0.f;
      if (rng_uniform(i1, 7u) >= 0.9f) p1 = 0.f;
      Ps[wave][row * AT_KC + (lane & 15)]      = f32_to_bf16(p0);
      Ps[wave][row * AT_KC + 16 + (lane & 15)] = f32_to_bf16(p1);
    }

    // P (16x32, A layout from per-wave LDS scratch)  x  V chunk (32x64)
    v16bf pa = load_A(&Ps[wave][0], AT_KC, lane);
#pragma unroll
    for (int tn = 0; tn < 4; ++tn) {
      v16bf vb = load_Bt(Vt + (tn * 16) * VTS, VTS, lane);
      o[tn] = wmma_bf16(pa, vb, o[tn]);
    }
  }

  // normalize (fold inverted-dropout 1/(1-p)) and store [B,S,DMODEL] bf16
  const int b_ = bh >> 4, h_ = bh & 15;
#pragma unroll
  for (int j = 0; j < 8; ++j) {
    int row = j + 8 * (lane >> 4);
    float inv = (lrow[j] > 0.f) ? 1.f / (0.9f * lrow[j]) : 0.f;
    size_t base = ((size_t)(b_ * S_LEN + q0 + row)) * DMODEL + h_ * DK;
#pragma unroll
    for (int tn = 0; tn < 4; ++tn)
      AO[base + tn * 16 + (lane & 15)] = f32_to_bf16(o[tn][j] * inv);
  }
}

// ---------------------------------------------------------------------------
// Host-side launcher
// ---------------------------------------------------------------------------
extern "C" void kernel_launch(void* const* d_in, const int* in_sizes, int n_in,
                              void* d_out, int out_size, void* d_ws, size_t ws_size,
                              hipStream_t stream) {
  const float* x_q = (const float*)d_in[0];
  const float* x_k = (const float*)d_in[1];
  const float* x_v = (const float*)d_in[2];
  const float* Wq = (const float*)d_in[3];  const float* bq = (const float*)d_in[4];
  const float* Wk = (const float*)d_in[5];  const float* bk = (const float*)d_in[6];
  const float* Wv = (const float*)d_in[7];  const float* bv = (const float*)d_in[8];
  const float* Wo = (const float*)d_in[9];  const float* bo = (const float*)d_in[10];

  unsigned char* ws = (unsigned char*)d_ws;
  const size_t U  = (size_t)MROWS * DMODEL * 2;   // 16 MiB: one [8192,1024] bf16
  const size_t WU = (size_t)DMODEL * DMODEL * 2;  //  2 MiB: one [1024,1024] bf16
  unsigned short* Xq  = (unsigned short*)(ws + 0 * U);
  unsigned short* Xk  = (unsigned short*)(ws + 1 * U);
  unsigned short* Xv  = (unsigned short*)(ws + 2 * U);
  unsigned short* Wqb = (unsigned short*)(ws + 3 * U + 0 * WU);
  unsigned short* Wkb = (unsigned short*)(ws + 3 * U + 1 * WU);
  unsigned short* Wvb = (unsigned short*)(ws + 3 * U + 2 * WU);
  unsigned short* Wob = (unsigned short*)(ws + 3 * U + 3 * WU);
  unsigned short* Qp  = (unsigned short*)(ws + 3 * U + 4 * WU + 0 * U);
  unsigned short* Kp  = (unsigned short*)(ws + 3 * U + 4 * WU + 1 * U);
  unsigned short* Vp  = (unsigned short*)(ws + 3 * U + 4 * WU + 2 * U);
  unsigned short* AO  = (unsigned short*)(ws + 3 * U + 4 * WU + 3 * U);

  const int NX = MROWS * DMODEL;     // 8,388,608
  const int NW = DMODEL * DMODEL;    // 1,048,576
  cvt_bf16<<<4096, 256, 0, stream>>>(x_q, Xq, NX);
  cvt_bf16<<<4096, 256, 0, stream>>>(x_k, Xk, NX);
  cvt_bf16<<<4096, 256, 0, stream>>>(x_v, Xv, NX);
  cvt_bf16<<<2048, 256, 0, stream>>>(Wq, Wqb, NW);
  cvt_bf16<<<2048, 256, 0, stream>>>(Wk, Wkb, NW);
  cvt_bf16<<<2048, 256, 0, stream>>>(Wv, Wvb, NW);
  cvt_bf16<<<2048, 256, 0, stream>>>(Wo, Wob, NW);

  dim3 gg(MROWS / GM_MBLK, DMODEL / GM_NBLK);  // (64,16)
  // QKV projections (Q gets 1/sqrt(d_k)=0.125 folded in; bias added pre-scale per ref)
  gemm_bf16<<<gg, 256, 0, stream>>>(Xq, Wqb, bq, Qp, nullptr, MROWS, DMODEL, DMODEL, 0.125f, 0);
  gemm_bf16<<<gg, 256, 0, stream>>>(Xk, Wkb, bk, Kp, nullptr, MROWS, DMODEL, DMODEL, 1.0f, 0);
  gemm_bf16<<<gg, 256, 0, stream>>>(Xv, Wvb, bv, Vp, nullptr, MROWS, DMODEL, DMODEL, 1.0f, 0);

  attn_kernel<<<dim3(BHEADS, S_LEN / 128), 256, 0, stream>>>(Qp, Kp, Vp, AO);

  // output projection -> f32 d_out
  gemm_bf16<<<gg, 256, 0, stream>>>(AO, Wob, bo, nullptr, (float*)d_out,
                                    MROWS, DMODEL, DMODEL, 1.0f, 1);
}
